// AttentionGRU_10634339025020
// MI455X (gfx1250) — compile-verified, matrix operand order checked by reading, not run on hardware
//
#include <hip/hip_runtime.h>

// ---------------------------------------------------------------------------
// AttentionGRU for MI455X (gfx1250, wave32, WMMA, TDM).
// B=32, S=512, H=1024.
//  1) q = query@W^T + W_b ; p = q@U  (U-GEMM folded; q.U_b cancels in softmax)
//  2) scores[b,s] = memory[b,s,:].p[b,:] (+mask) -> softmax -> g[s,b]
//  3) x_{r,n}^{f,b} = memory@{f_ir,f_in,b_ir,b_in}^T + bias  (bf16 WMMA,
//     137 GFLOP, bf16 output to halve HBM traffic: 128MB vs 256MB)
//  4) recurrence: ONE persistent kernel, 64 blocks x 128 threads, software
//     grid barrier per step. Each block TDM-loads its 4 recurrent weight
//     tiles (16 cols x 1024 K bf16, LDS-padded) into LDS ONCE, then all 512
//     steps read B-operands from LDS -> removes the 16MB/step L2 weight
//     stream and 511 launch overheads. h kept f32 + bf16 mirror.
// ---------------------------------------------------------------------------

#define DEVFN __device__ __forceinline__

typedef __bf16 bf16_t;
typedef __attribute__((ext_vector_type(16))) __bf16 v16bf;
typedef __attribute__((ext_vector_type(8)))  __bf16 v8bf;
typedef __attribute__((ext_vector_type(8)))  float  v8f;
typedef unsigned int u32x4 __attribute__((ext_vector_type(4)));
typedef int i32x8 __attribute__((ext_vector_type(8)));
typedef int i32x4 __attribute__((ext_vector_type(4)));

constexpr int B = 32, S = 512, H = 1024;
constexpr int NB = 64;                 // persistent-kernel block count
// LDS weight tile: 16 rows x 1024 bf16, +16B pad every 1024B (TDM pad)
constexpr int ROWB = 2080;             // padded row stride in bytes
constexpr int SECB = 16 * ROWB;        // one weight tile section = 33280 B
constexpr int LDSB = 4 * SECB;         // 4 sections per block = 133120 B

DEVFN v8f wmma_bf16(v16bf a, v16bf b, v8f c) {
  return __builtin_amdgcn_wmma_f32_16x16x32_bf16(false, a, false, b, (short)0, c,
                                                 false, false);
}

DEVFN v16bf cat8(v8bf lo, v8bf hi) {
  v16bf r;
#pragma unroll
  for (int e = 0; e < 8; ++e) { r[e] = lo[e]; r[8 + e] = hi[e]; }
  return r;
}

// A operand: rows of bf16 matrix (row-major, leading dim H).
// Lane layout: M = lane&15; K chunks {k0+half*8 .. +7} and {+16..+23}.
DEVFN v16bf loadA_rows(const bf16_t* __restrict__ base, int row0, int k0, int lane) {
  const bf16_t* src =
      base + (size_t)(row0 + (lane & 15)) * H + k0 + ((lane >> 4) << 3);
  return cat8(*(const v8bf*)src, *(const v8bf*)(src + 16));
}

// A operand rows from memory laid out [B][S][H]; logical GEMM row r = s*B+b.
DEVFN v16bf loadA_mem(const bf16_t* __restrict__ mem, int row0, int k0, int lane) {
  int r = row0 + (lane & 15);
  int b = r & (B - 1);
  int s = r >> 5;
  const bf16_t* src =
      mem + ((size_t)b * S + s) * H + k0 + ((lane >> 4) << 3);
  return cat8(*(const v8bf*)src, *(const v8bf*)(src + 16));
}

// B operand from global weights W (row-major [n][k]); lane&15 = N.
DEVFN v16bf loadB_g(const bf16_t* __restrict__ w, int n0, int k0, int lane) {
  const bf16_t* src =
      w + (size_t)(n0 + (lane & 15)) * H + k0 + ((lane >> 4) << 3);
  return cat8(*(const v8bf*)src, *(const v8bf*)(src + 16));
}

// B operand from an LDS-resident padded tile (16 rows x 1024 K bf16).
DEVFN v16bf loadB_lds(const char* sec, int k0, int lane) {
  int row = lane & 15;
  int kk = k0 + ((lane >> 4) << 3);
  int k2 = kk + 16;
  int o1 = row * ROWB + kk * 2 + ((kk >> 9) << 4);
  int o2 = row * ROWB + k2 * 2 + ((k2 >> 9) << 4);
  return cat8(*(const v8bf*)(sec + o1), *(const v8bf*)(sec + o2));
}

// ------------------------------- small kernels ------------------------------

__global__ void __launch_bounds__(256) cvt_f32_bf16(const float* __restrict__ src,
                                                    bf16_t* __restrict__ dst, int n) {
  for (int i = blockIdx.x * blockDim.x + threadIdx.x; i < n;
       i += gridDim.x * blockDim.x)
    dst[i] = (bf16_t)src[i];
}

__global__ void __launch_bounds__(256) qproj(const float* __restrict__ query,
                                             const float* __restrict__ Ww,
                                             const float* __restrict__ Wb,
                                             float* __restrict__ q) {
  int idx = blockIdx.x * 256 + threadIdx.x;
  if (idx >= B * H) return;
  int b = idx >> 10, h = idx & (H - 1);
  const float* qr = query + (size_t)b * H;
  const float* wr = Ww + (size_t)h * H;
  float acc = Wb[h];
  for (int i = 0; i < H; ++i) acc += qr[i] * wr[i];
  q[idx] = acc;
}

__global__ void __launch_bounds__(256) pproj(const float* __restrict__ q,
                                             const float* __restrict__ Uw,
                                             float* __restrict__ p) {
  int idx = blockIdx.x * 256 + threadIdx.x;
  if (idx >= B * H) return;
  int b = idx >> 10, i = idx & (H - 1);
  float acc = 0.0f;
  for (int h = 0; h < H; ++h) acc += q[(size_t)b * H + h] * Uw[(size_t)h * H + i];
  p[idx] = acc;
}

__global__ void __launch_bounds__(256) attn_scores(const float* __restrict__ mem,
                                                   const float* __restrict__ p,
                                                   const unsigned char* __restrict__ mask,
                                                   float* __restrict__ scores) {
  int wid = (blockIdx.x * 256 + threadIdx.x) >> 5;
  int lane = threadIdx.x & 31;
  int b = wid >> 9, s = wid & (S - 1);
  const float* mrow = mem + ((size_t)b * S + s) * H;
  const float* pr = p + (size_t)b * H;
  float acc = 0.0f;
  for (int i = lane; i < H; i += 32) acc += mrow[i] * pr[i];
#pragma unroll
  for (int o = 16; o > 0; o >>= 1) acc += __shfl_down(acc, o, 32);
  if (lane == 0)
    scores[wid] = mask[(size_t)b * S + s] ? -__builtin_huge_valf() : acc;
}

__global__ void __launch_bounds__(512) attn_softmax(const float* __restrict__ scores,
                                                    float* __restrict__ g) {
  __shared__ float red[512];
  int b = blockIdx.x, t = threadIdx.x;
  float v = scores[(size_t)b * S + t];
  red[t] = v;
  __syncthreads();
  for (int st = 256; st > 0; st >>= 1) {
    if (t < st) red[t] = fmaxf(red[t], red[t + st]);
    __syncthreads();
  }
  float mx = red[0];
  __syncthreads();
  float e = __expf(v - mx);
  red[t] = e;
  __syncthreads();
  for (int st = 256; st > 0; st >>= 1) {
    if (t < st) red[t] = red[t] + red[t + st];
    __syncthreads();
  }
  g[(size_t)t * B + b] = e / red[0];
}

// ------------------------- big input-projection GEMM ------------------------
// out[s*B+b, j] = (bf16)(sum_i mem[b,s,i]*w[j,i] + bias[j]); M=16384,N=K=1024.
__global__ void __launch_bounds__(256) proj_gemm(const bf16_t* __restrict__ mem,
                                                 const bf16_t* __restrict__ w,
                                                 const float* __restrict__ bias,
                                                 bf16_t* __restrict__ out) {
  int lane = threadIdx.x & 31;
  int wid = blockIdx.x * 8 + (threadIdx.x >> 5);  // 0..16383
  int nt = wid & 31, mt = wid >> 5;
  int row0 = mt * 32, n0 = nt * 32;
  v8f acc00 = {}, acc01 = {}, acc10 = {}, acc11 = {};
  for (int k0 = 0; k0 < H; k0 += 32) {
    v16bf a0 = loadA_mem(mem, row0, k0, lane);
    v16bf a1 = loadA_mem(mem, row0 + 16, k0, lane);
    v16bf b0 = loadB_g(w, n0, k0, lane);
    v16bf b1 = loadB_g(w, n0 + 16, k0, lane);
    acc00 = wmma_bf16(a0, b0, acc00);
    acc01 = wmma_bf16(a0, b1, acc01);
    acc10 = wmma_bf16(a1, b0, acc10);
    acc11 = wmma_bf16(a1, b1, acc11);
  }
  int Nl = lane & 15, Mo = (lane >> 4) << 3;
#pragma unroll
  for (int r = 0; r < 8; ++r) {
    int m0r = row0 + Mo + r, m1r = m0r + 16;
    int na = n0 + Nl, nb = na + 16;
    out[(size_t)m0r * H + na] = (bf16_t)(acc00[r] + bias[na]);
    out[(size_t)m0r * H + nb] = (bf16_t)(acc01[r] + bias[nb]);
    out[(size_t)m1r * H + na] = (bf16_t)(acc10[r] + bias[na]);
    out[(size_t)m1r * H + nb] = (bf16_t)(acc11[r] + bias[nb]);
  }
}

// --------------------- persistent recurrent scan kernel ---------------------

DEVFN void grid_barrier(unsigned* cnt, unsigned target) {
  __threadfence();
  __syncthreads();
  if (threadIdx.x == 0) {
    atomicAdd(cnt, 1u);
    volatile unsigned* vc = cnt;
    while (*vc < target) __builtin_amdgcn_s_sleep(1);
  }
  __syncthreads();
  __threadfence();
}

// Issue one TDM load: 2D tile of 16 rows x 1024 bf16 from row-major weights
// (leading dim H) into LDS at byte offset ldsOff, 16B pad every 1024B.
// 6-arg builtin form: (g0:u32x4, g1:i32x8, g2:i32x4, g3:i32x4, x:i32x8, cpol).
DEVFN void tdm_load_tile(const bf16_t* gptr, unsigned ldsOff) {
  unsigned long long ga = (unsigned long long)(uintptr_t)gptr;
  u32x4 g0;
  g0.x = 1u;                                     // count=1, no gather
  g0.y = ldsOff;                                 // lds_addr (bytes)
  g0.z = (unsigned)(ga & 0xFFFFFFFFu);           // global_addr[31:0]
  g0.w = (unsigned)((ga >> 32) & 0x01FFFFFFu) | 0x80000000u;  // [56:32]|type=2
  i32x8 g1;
  g1[0] = (int)((1u << 16)      // data_size = 2B
              | (1u << 20)      // pad_enable
              | (7u << 22)      // pad_interval: 256 DWORDs = 1024B
              | (3u << 25));    // pad_amount: 4 DWORDs = 16B
  g1[1] = (int)((1024u & 0xFFFFu) << 16);                 // tensor_dim0 lo16
  g1[2] = (int)(((1024u >> 16) & 0xFFFFu) | (1024u << 16)); // dim0 hi | dim1 lo
  g1[3] = (int)(((1024u >> 16) & 0xFFFFu) | (1024u << 16)); // dim1 hi | tile_dim0
  g1[4] = 16;                                             // tile_dim1=16
  g1[5] = 1024;                                           // tensor_dim0_stride
  g1[6] = 0;
  g1[7] = 0;
  i32x4 gz = { 0, 0, 0, 0 };
  i32x8 gz8 = { 0, 0, 0, 0, 0, 0, 0, 0 };
  __builtin_amdgcn_tensor_load_to_lds(g0, g1, gz, gz, gz8, 0);
}

// 64 blocks x 128 threads (4 waves). Wave w: dir = w>>1, mt = w&1; nt = bid.
// LDS sections [dir*2+mat]: weight tile cols nt*16..+16, all K (bf16, padded).
__global__ void __launch_bounds__(128) gru_scan(
    const float* __restrict__ hidden,
    float* __restrict__ hA, float* __restrict__ hB,
    bf16_t* __restrict__ hbA, bf16_t* __restrict__ hbB,
    const bf16_t* __restrict__ wrec,   // [4][H*H]: f_hr, f_hn, b_hr, b_hn
    const float* __restrict__ f_hr_b, const float* __restrict__ f_hn_b,
    const float* __restrict__ b_hr_b, const float* __restrict__ b_hn_b,
    const bf16_t* __restrict__ xs,     // [4][S*B*H]: xr_f, xn_f, xr_b, xn_b
    const float* __restrict__ g,
    float* __restrict__ out, unsigned* __restrict__ cnt) {
  extern __shared__ char smem[];
  int tid = threadIdx.x, lane = tid & 31, w = tid >> 5;
  int dir = w >> 1, mt = w & 1, nt = blockIdx.x;
  const size_t HH = (size_t)H * H;
  const size_t SBH = (size_t)S * B * H;

  // --- stage this block's 4 recurrent weight tiles into LDS via TDM ---
  if (w == 0) {
    if (lane == 0) {
#pragma unroll
      for (int m = 0; m < 4; ++m)
        tdm_load_tile(wrec + (size_t)m * HH + (size_t)nt * 16 * H,
                      (unsigned)(m * SECB));
    }
    __builtin_amdgcn_s_wait_tensorcnt(0);
  }

  // --- init h state (f32 + bf16 mirror) from `hidden` ---
  for (int i = blockIdx.x * 128 + tid; i < 2 * B * H; i += NB * 128) {
    float v = hidden[i];
    hA[i] = v;
    hbA[i] = (bf16_t)v;
  }
  unsigned target = NB;
  grid_barrier(cnt, target);

  const char* secR = smem + (size_t)(dir * 2 + 0) * SECB;
  const char* secN = smem + (size_t)(dir * 2 + 1) * SECB;
  const float* brb = dir ? b_hr_b : f_hr_b;
  const float* bnb = dir ? b_hn_b : f_hn_b;
  const bf16_t* xr = xs + (size_t)(dir * 2) * SBH;
  const bf16_t* xn = xs + (size_t)(dir * 2 + 1) * SBH;
  int row0 = mt * 16, n0 = nt * 16;
  int Nl = lane & 15, Mo = (lane >> 4) << 3;
  int j = n0 + Nl;

  for (int step = 0; step < S; ++step) {
    const float*  hf_in  = (step & 1) ? hB : hA;
    float*        hf_out = (step == S - 1) ? out : ((step & 1) ? hA : hB);
    const bf16_t* hb_in  = (step & 1) ? hbB : hbA;
    bf16_t*       hb_out = (step & 1) ? hbA : hbB;
    const float*  hd  = hf_in + (size_t)dir * B * H;
    const bf16_t* hbd = hb_in + (size_t)dir * B * H;
    float*        hfo = hf_out + (size_t)dir * B * H;
    bf16_t*       hbo = hb_out + (size_t)dir * B * H;
    int s_idx = dir ? (S - 1 - step) : step;

    v8f ar = {}, an = {};
    for (int k0 = 0; k0 < H; k0 += 32) {
      v16bf a  = loadA_rows(hbd, row0, k0, lane);
      v16bf b0 = loadB_lds(secR, k0, lane);
      v16bf b1 = loadB_lds(secN, k0, lane);
      ar = wmma_bf16(a, b0, ar);
      an = wmma_bf16(a, b1, an);
    }
#pragma unroll
    for (int r = 0; r < 8; ++r) {
      int brow = row0 + Mo + r;
      float hr = ar[r] + brb[j];
      float hn = an[r] + bnb[j];
      size_t xi = ((size_t)s_idx * B + brow) * H + j;
      float rr = 1.0f / (1.0f + __expf(-((float)xr[xi] + hr)));
      float nnv = tanhf((float)xn[xi] + rr * hn);
      float gv = g[(size_t)s_idx * B + brow];
      float hprev = hd[(size_t)brow * H + j];
      float hnew = gv * nnv + (1.0f - gv) * hprev;
      hfo[(size_t)brow * H + j] = hnew;
      hbo[(size_t)brow * H + j] = (bf16_t)hnew;
    }
    if (step != S - 1) {
      target += NB;
      grid_barrier(cnt, target);
    }
  }
}

// --------------------------------- launcher ---------------------------------

extern "C" void kernel_launch(void* const* d_in, const int* in_sizes, int n_in,
                              void* d_out, int out_size, void* d_ws, size_t ws_size,
                              hipStream_t stream) {
  (void)in_sizes; (void)n_in; (void)out_size; (void)ws_size;
  const float* query  = (const float*)d_in[0];
  const float* memory = (const float*)d_in[1];
  const float* hidden = (const float*)d_in[2];
  const unsigned char* mask = (const unsigned char*)d_in[3];
  const float* W_w = (const float*)d_in[4];
  const float* W_b = (const float*)d_in[5];
  const float* U_w = (const float*)d_in[6];
  // d_in[7] = U_b: per-batch-constant in scores -> cancelled by softmax.
  const float* f_ir_w = (const float*)d_in[8];
  const float* f_ir_b = (const float*)d_in[9];
  const float* f_hr_w = (const float*)d_in[10];
  const float* f_hr_b = (const float*)d_in[11];
  const float* f_in_w = (const float*)d_in[12];
  const float* f_in_b = (const float*)d_in[13];
  const float* f_hn_w = (const float*)d_in[14];
  const float* f_hn_b = (const float*)d_in[15];
  const float* b_ir_w = (const float*)d_in[16];
  const float* b_ir_b = (const float*)d_in[17];
  const float* b_hr_w = (const float*)d_in[18];
  const float* b_hr_b = (const float*)d_in[19];
  const float* b_in_w = (const float*)d_in[20];
  const float* b_in_b = (const float*)d_in[21];
  const float* b_hn_w = (const float*)d_in[22];
  const float* b_hn_b = (const float*)d_in[23];

  char* ws = (char*)d_ws;
  size_t off = 0;
  auto carve = [&](size_t bytes) -> void* {
    void* pp = ws + off;
    off += (bytes + 255) & ~(size_t)255;
    return pp;
  };
  const size_t HH = (size_t)H * H;
  const size_t SBH = (size_t)S * B * H;
  bf16_t* mem_bf = (bf16_t*)carve((size_t)B * S * H * sizeof(bf16_t));
  bf16_t* wproj  = (bf16_t*)carve(4 * HH * sizeof(bf16_t)); // f_ir,f_in,b_ir,b_in
  bf16_t* wrec   = (bf16_t*)carve(4 * HH * sizeof(bf16_t)); // f_hr,f_hn,b_hr,b_hn
  float* qv      = (float*)carve((size_t)B * H * sizeof(float));
  float* pv      = (float*)carve((size_t)B * H * sizeof(float));
  float* scores  = (float*)carve((size_t)B * S * sizeof(float));
  float* gbuf    = (float*)carve((size_t)S * B * sizeof(float));
  bf16_t* xs     = (bf16_t*)carve(4 * SBH * sizeof(bf16_t)); // xr_f,xn_f,xr_b,xn_b
  float* hA      = (float*)carve((size_t)2 * B * H * sizeof(float));
  float* hB      = (float*)carve((size_t)2 * B * H * sizeof(float));
  bf16_t* hbA    = (bf16_t*)carve((size_t)2 * B * H * sizeof(bf16_t));
  bf16_t* hbB    = (bf16_t*)carve((size_t)2 * B * H * sizeof(bf16_t));
  unsigned* cnt  = (unsigned*)carve(256);

  // bf16 conversions
  cvt_f32_bf16<<<8192, 256, 0, stream>>>(memory, mem_bf, B * S * H);
  const float* projw[4] = { f_ir_w, f_in_w, b_ir_w, b_in_w };
  const float* recw[4]  = { f_hr_w, f_hn_w, b_hr_w, b_hn_w };
  for (int i = 0; i < 4; ++i) {
    cvt_f32_bf16<<<2048, 256, 0, stream>>>(projw[i], wproj + i * HH, (int)HH);
    cvt_f32_bf16<<<2048, 256, 0, stream>>>(recw[i],  wrec  + i * HH, (int)HH);
  }

  // attention gates
  qproj<<<(B * H) / 256, 256, 0, stream>>>(query, W_w, W_b, qv);
  pproj<<<(B * H) / 256, 256, 0, stream>>>(qv, U_w, pv);
  attn_scores<<<(B * S * 32) / 256, 256, 0, stream>>>(memory, pv, mask, scores);
  attn_softmax<<<B, 512, 0, stream>>>(scores, gbuf);

  // input projections (WMMA GEMMs), bf16 output
  const float* projb[4] = { f_ir_b, f_in_b, b_ir_b, b_in_b };
  for (int i = 0; i < 4; ++i)
    proj_gemm<<<2048, 256, 0, stream>>>(mem_bf, wproj + i * HH, projb[i],
                                        xs + (size_t)i * SBH);

  // persistent recurrent scan: one launch, grid barrier per step
  (void)hipMemsetAsync(cnt, 0, sizeof(unsigned), stream);
  gru_scan<<<NB, 128, LDSB, stream>>>(hidden, hA, hB, hbA, hbB, wrec,
                                      f_hr_b, f_hn_b, b_hr_b, b_hn_b,
                                      xs, gbuf, (float*)d_out, cnt);
}
